// lmy_26139170964251
// MI455X (gfx1250) — matrix operand order repaired and checked by reading, not compiled, not run on hardware
//
#include <hip/hip_runtime.h>
#include <hip/hip_bf16.h>

// Shapes from the reference
constexpr int B_   = 16;
constexpr int H_   = 64;
constexpr int W_   = 64;
constexpr int CIN  = 64;
constexpr int COUT = 128;
constexpr int NS   = 8;   // NSAMPLE

typedef __attribute__((ext_vector_type(16))) __bf16 v16bf;
typedef __attribute__((ext_vector_type(8)))  __bf16 v8bf;
typedef __attribute__((ext_vector_type(4)))  __bf16 v4bf;
typedef __attribute__((ext_vector_type(8)))  float  v8f;

// ---------------- prep 1: x fp32 -> bf16 (same NHWC layout) ----------------
// x is read exactly once -> non-temporal loads keep it out of L2; the bf16
// copy (8 MB) is the hot, L2-resident version.
__global__ void cvt_x_kernel(const float* __restrict__ x, __bf16* __restrict__ xbf) {
    int i = (blockIdx.x * blockDim.x + threadIdx.x) * 4;   // 4,194,304 elems total
    float4 v;
    v.x = __builtin_nontemporal_load(x + i + 0);
    v.y = __builtin_nontemporal_load(x + i + 1);
    v.z = __builtin_nontemporal_load(x + i + 2);
    v.w = __builtin_nontemporal_load(x + i + 3);
    v4bf o;
    o[0] = (__bf16)v.x; o[1] = (__bf16)v.y; o[2] = (__bf16)v.z; o[3] = (__bf16)v.w;
    *(v4bf*)(xbf + i) = o;
}

// ------- prep 2: sample kernels + transpose to [s, khw, cout, cin] bf16 -------
__global__ void sample_kernels(const float* __restrict__ mu,
                               const float* __restrict__ logstd,
                               const float* __restrict__ noise,
                               __bf16* __restrict__ kT) {
    int idx = blockIdx.x * blockDim.x + threadIdx.x;       // 8*9*64*128 = 589824
    if (idx >= NS * 9 * CIN * COUT) return;
    int s    = idx / (9 * CIN * COUT);
    int rem  = idx - s * (9 * CIN * COUT);                 // index into mu/logstd
    int khw  = rem / (CIN * COUT);
    int r2   = rem - khw * (CIN * COUT);
    int cin  = r2 / COUT;
    int cout = r2 - cin * COUT;
    float v = mu[rem] + __builtin_nontemporal_load(noise + idx) * __expf(logstd[rem]);
    kT[((size_t)(s * 9 + khw) * COUT + cout) * CIN + cin] = (__bf16)v;
}

// ---------------- main: implicit-GEMM conv via bf16 WMMA ----------------
// grid: (H, B, NS), block: 256 threads = 8 waves.
// Block tile: M = 64 (full W row) x N = 128 couts for one (s,b,h).
// Wave tile : 32x32 -> 4 accumulators; every A/B fragment feeds 2 WMMAs.
// K loop    : 3x3 taps x 2 cin-chunks of 32 = 18 steps x 4 WMMA = 72 WMMA/wave.
__global__ __launch_bounds__(256, 2)
void conv_wmma(const __bf16* __restrict__ xbf,
               const __bf16* __restrict__ kT,
               float* __restrict__ out) {
    __shared__ __bf16 lds[3 * 66 * CIN];                   // 25,344 B slab w/ halo

    const int h   = blockIdx.x;
    const int b   = blockIdx.y;
    const int s   = blockIdx.z;
    const int tid = threadIdx.x;

    // ---- Stage x slab: rows hh = h-1..h+1, cols ww = -1..64, 64 cin. ----
    // 3*66*8 = 1584 chunks of 8 bf16 (16 B).
    // Valid chunks: CDNA5 async copy global->LDS (ASYNCcnt). OOB: zero ds_store.
    // Low 32 bits of the flat shared address are the LDS byte offset.
    const unsigned lds_base = (unsigned)(uintptr_t)(&lds[0]);
    for (int ch = tid; ch < 1584; ch += 256) {
        int k8 = ch & 7;                  // cin chunk (8 bf16)
        int rc = ch >> 3;                 // 0..197
        int c  = rc % 66;                 // ww = c - 1
        int r  = rc / 66;                 // hh = h - 1 + r
        int hh = h - 1 + r;
        int ww = c - 1;
        unsigned lds_off = lds_base + ch * 16;
        if ((unsigned)hh < (unsigned)H_ && (unsigned)ww < (unsigned)W_) {
            const __bf16* gp = xbf + ((size_t)((b * H_ + hh) * W_ + ww) * CIN) + k8 * 8;
            asm volatile("global_load_async_to_lds_b128 %0, %1, off"
                         :: "v"(lds_off), "v"(gp) : "memory");
        } else {
            v8bf z = {};
            *(v8bf*)(lds + ch * 8) = z;
        }
    }
    asm volatile("s_wait_asynccnt 0x0" ::: "memory");
    __syncthreads();

    const int lane = tid & 31;
    const int wv   = tid >> 5;            // 0..7
    const int wr   = wv >> 2;             // wave M-row (0..1)  -> M base wr*32
    const int wc   = wv & 3;              // wave N-col (0..3)  -> N base wc*32
    const int m    = lane & 15;           // A: M row / B,D: N col (within 16)
    const int hl   = lane >> 4;           // lane-half selects K sub-ranges

    v8f acc[2][2] = {};

    #pragma unroll
    for (int kh = 0; kh < 3; ++kh) {
        #pragma unroll
        for (int kw = 0; kw < 3; ++kw) {
            const int tap = kh * 3 + kw;
            #pragma unroll
            for (int kc = 0; kc < 2; ++kc) {
                // A fragments (LDS): 16x32 bf16 layout:
                // lanes 0-15 hold K {0..7,16..23}, lanes 16-31 hold K {8..15,24..31}
                v16bf a[2];
                #pragma unroll
                for (int mi = 0; mi < 2; ++mi) {
                    const __bf16* arow = lds
                        + (kh * 66 + kw + wr * 32 + mi * 16 + m) * CIN
                        + kc * 32 + hl * 8;
                    v8bf alo = *(const v8bf*)(arow);
                    v8bf ahi = *(const v8bf*)(arow + 16);
                    a[mi] = __builtin_shufflevector(alo, ahi,
                              0,1,2,3,4,5,6,7,8,9,10,11,12,13,14,15);
                }
                // B fragments (L2-resident kT): lane = N, K {0..7,16..23} / {8..15,24..31} per half
                v16bf bm[2];
                #pragma unroll
                for (int ni = 0; ni < 2; ++ni) {
                    const __bf16* brow = kT
                        + ((size_t)(s * 9 + tap) * COUT + wc * 32 + ni * 16 + m) * CIN
                        + kc * 32 + hl * 8;
                    v8bf blo = *(const v8bf*)(brow);
                    v8bf bhi = *(const v8bf*)(brow + 16);
                    bm[ni] = __builtin_shufflevector(blo, bhi,
                              0,1,2,3,4,5,6,7,8,9,10,11,12,13,14,15);
                }
                #pragma unroll
                for (int mi = 0; mi < 2; ++mi)
                    #pragma unroll
                    for (int ni = 0; ni < 2; ++ni)
                        acc[mi][ni] = __builtin_amdgcn_wmma_f32_16x16x32_bf16(
                            false, a[mi], false, bm[ni], (short)0, acc[mi][ni],
                            false, false);
            }
        }
    }

    // ---- ELU + store (non-temporal: 256 MB write-once stream must not evict
    // the L2-resident xbf/kT working set).
    // D layout: VGPR r -> M=r (+8 for lanes 16-31), lane&15 -> N.
    constexpr int ROWSTRIDE = NS * COUT;  // stride between consecutive w positions
    #pragma unroll
    for (int mi = 0; mi < 2; ++mi) {
        #pragma unroll
        for (int ni = 0; ni < 2; ++ni) {
            float* op = out
                + ((size_t)(((b * H_ + h) * W_ + wr * 32 + mi * 16) * NS + s) * COUT)
                + wc * 32 + ni * 16 + m;
            #pragma unroll
            for (int r = 0; r < 8; ++r) {
                float v = acc[mi][ni][r];
                v = v > 0.0f ? v : (__expf(v) - 1.0f);
                __builtin_nontemporal_store(v, op + (size_t)(r + hl * 8) * ROWSTRIDE);
            }
        }
    }
}

extern "C" void kernel_launch(void* const* d_in, const int* in_sizes, int n_in,
                              void* d_out, int out_size, void* d_ws, size_t ws_size,
                              hipStream_t stream) {
    const float* x      = (const float*)d_in[0];
    const float* mu     = (const float*)d_in[1];
    const float* logstd = (const float*)d_in[2];
    const float* noise  = (const float*)d_in[3];

    // workspace: x in bf16 (8 MB) then transposed sampled kernels (1.125 MB)
    __bf16* xbf = (__bf16*)d_ws;
    __bf16* kT  = (__bf16*)((char*)d_ws + (size_t)B_ * H_ * W_ * CIN * sizeof(__bf16));

    cvt_x_kernel<<<4096, 256, 0, stream>>>(x, xbf);
    sample_kernels<<<(NS * 9 * CIN * COUT + 255) / 256, 256, 0, stream>>>(mu, logstd, noise, kT);

    dim3 grid(H_, B_, NS);                // 64 x 16 x 8 = 8192 blocks
    conv_wmma<<<grid, 256, 0, stream>>>(xbf, kT, (float*)d_out);
}
